// GraphAttention_32177894981958
// MI455X (gfx1250) — matrix operand order repaired and checked by reading, not compiled, not run on hardware
//
#include <hip/hip_runtime.h>
#include <hip/hip_bf16.h>

// ---------------- problem constants ----------------
constexpr int nNodes = 10000;
constexpr int nEdges = 320000;
constexpr int S    = 128;   // S_IN
constexpr int V    = 64;    // V_IN
constexpr int ATTR = 64;
constexpr int NB   = 32;
constexpr int HEADS = 4;
constexpr int SHD  = 32;    // SH
constexpr int FC   = 64;
constexpr int STP  = 192;   // S_TP
constexpr int VTP  = 256;   // V_TP
constexpr int NW   = 448;
constexpr int FIN  = 160;   // NB + 2*ATTR
constexpr int PAD  = 4;     // padding for f32 scratch rows
constexpr int PADB = 8;     // padding for bf16 GEMM-A rows (keeps rows 16B aligned)

typedef __attribute__((ext_vector_type(16))) __bf16 v16bf;
typedef __attribute__((ext_vector_type(8)))  __bf16 v8bf;
typedef __attribute__((ext_vector_type(8)))  float  v8f;

__device__ __forceinline__ float sigf(float x) { return 1.0f / (1.0f + expf(-x)); }

// Wave-level 16x16 GEMM tile: C(16x16) += A(16xK bf16 LDS, row stride lda) *
// B^T (bf16 global, [N][Kb] transposed weights, column block col0).
// wave32 WMMA layouts used here (contiguous k order chosen by this code, same for A and B):
//   A frag: row m = lane&15 ; elem e -> k = (lane>=16?16:0) + e
//           -> two contiguous 8-element (16B) chunks per lane.
//   B frag: col n = lane&15 ; elem e -> k = (lane>=16?16:0) + e
//           -> one contiguous 16-element (32B) chunk per lane in B^T.
//   C tile: reg r -> row = r + (lane>=16?8:0), col = lane&15
__device__ __forceinline__ v8f wmma_gemm16(const __bf16* A, int lda,
                                           const __bf16* Bt, int Kb, int col0,
                                           int K, v8f acc) {
  const int lane = threadIdx.x & 31;
  const int l15  = lane & 15;
  const int hf   = lane >> 4;
  const __bf16* arow = A + l15 * lda + (hf << 4);
  const __bf16* brow = Bt + (col0 + l15) * Kb + (hf << 4);
  for (int k0 = 0; k0 < K; k0 += 32) {
    const v8bf a0 = *(const v8bf*)(arow + k0);
    const v8bf a1 = *(const v8bf*)(arow + k0 + 8);
    const v16bf a = __builtin_shufflevector(a0, a1, 0, 1, 2, 3, 4, 5, 6, 7,
                                            8, 9, 10, 11, 12, 13, 14, 15);
    const v16bf b = *(const v16bf*)(brow + k0);
    acc = __builtin_amdgcn_wmma_f32_16x16x32_bf16(false, a, false, b,
                                                  (short)0, acc, false, false);
  }
  return acc;
}

__device__ __forceinline__ void atomicMaxFloat(float* addr, float val) {
  int old = __float_as_int(*addr);
  while (__int_as_float(old) < val) {
    const int assumed = old;
    old = atomicCAS((int*)addr, assumed, __float_as_int(val));
    if (old == assumed) break;
  }
}

// ---- one-time weight transpose+convert: src[K][N] f32 -> dst[N][K] bf16 ----
__global__ void wconv_kernel(const float* __restrict__ src,
                             __bf16* __restrict__ dst, int K, int N) {
  const int total = K * N;
  for (int i = blockIdx.x * blockDim.x + threadIdx.x; i < total;
       i += gridDim.x * blockDim.x) {
    const int k = i / N, n = i % N;
    dst[n * K + k] = (__bf16)src[i];
  }
}

// ---------------- node projections: ms, ds, mv, dv ----------------
__global__ __launch_bounds__(256) void node_proj_kernel(
    const float* __restrict__ node_input,
    const __bf16* __restrict__ tWss, const float* __restrict__ bss,
    const __bf16* __restrict__ tWsv,
    const __bf16* __restrict__ tWds, const __bf16* __restrict__ tWdv,
    float* __restrict__ ms, float* __restrict__ dsb,
    float* __restrict__ mv, float* __restrict__ dvv) {
  __shared__ __align__(16) __bf16 sS[16][S + PADB];
  __shared__ __align__(16) __bf16 sV[3][16][V + PADB];
  const int n0 = blockIdx.x * 16;
  for (int idx = threadIdx.x; idx < 16 * 320; idx += 256) {
    const int i = idx / 320, c = idx % 320;
    const float val = node_input[(n0 + i) * 320 + c];
    if (c < S) sS[i][c] = (__bf16)val;
    else { const int cc = c - S; sV[cc % 3][i][cc / 3] = (__bf16)val; }
  }
  __syncthreads();
  const int lane = threadIdx.x & 31, l15 = lane & 15, hf = lane >> 4;
  for (int job = threadIdx.x >> 5; job < 40; job += 8) {
    v8f acc = {};
    if (job < 8) {
      const int c0 = job * 16;
      acc = wmma_gemm16(&sS[0][0], S + PADB, tWss, S, c0, S, acc);
      const float b = bss[c0 + l15];
#pragma unroll
      for (int r = 0; r < 8; ++r)
        ms[(n0 + r + hf * 8) * S + c0 + l15] = acc[r] + b;
    } else if (job < 16) {
      const int c0 = (job - 8) * 16;
      acc = wmma_gemm16(&sS[0][0], S + PADB, tWds, S, c0, S, acc);
#pragma unroll
      for (int r = 0; r < 8; ++r)
        dsb[(n0 + r + hf * 8) * S + c0 + l15] = acc[r];
    } else if (job < 28) {
      const int j = job - 16, comp = j >> 2, c0 = (j & 3) * 16;
      acc = wmma_gemm16(&sV[comp][0][0], V + PADB, tWsv, V, c0, V, acc);
#pragma unroll
      for (int r = 0; r < 8; ++r)
        mv[(n0 + r + hf * 8) * (3 * V) + comp * V + c0 + l15] = acc[r];
    } else {
      const int j = job - 28, comp = j >> 2, c0 = (j & 3) * 16;
      acc = wmma_gemm16(&sV[comp][0][0], V + PADB, tWdv, V, c0, V, acc);
#pragma unroll
      for (int r = 0; r < 8; ++r)
        dvv[(n0 + r + hf * 8) * (3 * V) + comp * V + c0 + l15] = acc[r];
    }
  }
}

// ---------------- fused per-edge chain (two passes) ----------------
template <bool FULL>
__global__ __launch_bounds__(256) void edge_kernel(
    const float* __restrict__ ms, const float* __restrict__ dsb,
    const float* __restrict__ mv, const float* __restrict__ dvv,
    const int* __restrict__ edge_src, const int* __restrict__ edge_dst,
    const float* __restrict__ edge_attr, const float* __restrict__ edge_emb,
    const float* __restrict__ node_attr,
    const __bf16* __restrict__ tradW0, const float* __restrict__ radb0,
    const __bf16* __restrict__ tradW1, const float* __restrict__ radb1,
    const __bf16* __restrict__ tradW2, const float* __restrict__ radb2,
    const __bf16* __restrict__ tattWs, const float* __restrict__ attbs,
    const float* __restrict__ attdot,
    const __bf16* __restrict__ tlin1Ws, const float* __restrict__ lin1bs,
    const __bf16* __restrict__ tlin1Wv, const float* __restrict__ tp2w,
    const __bf16* __restrict__ tlin2Ws, const float* __restrict__ lin2bs,
    const __bf16* __restrict__ tlin2Wv,
    float* __restrict__ logits, const float* __restrict__ nmax,
    const float* __restrict__ nden,
    float* __restrict__ acc_s, float* __restrict__ acc_v) {
  __shared__ int    sSrc[16], sDst[16];
  __shared__ float  sY0[16], sY1[16][3];
  __shared__ float  sES[16][S + PAD];
  __shared__ float  sEV[16][V][3];
  __shared__ float  sEVD[16][V];
  __shared__ __align__(16) __bf16 sWF[16][FIN + PADB];
  __shared__ __align__(16) __bf16 sH1[16][FC + PADB];
  __shared__ __align__(16) __bf16 sH2[16][FC + PADB];
  __shared__ __align__(16) __bf16 sW[16][NW + PADB];
  __shared__ __align__(16) __bf16 sSTP[16][STP + PADB];
  __shared__ __align__(16) __bf16 sVTP[3][16][VTP + PADB];
  __shared__ float  sSC[16][S + PAD];
  __shared__ __align__(16) __bf16 sFS[16][S + PADB];
  __shared__ float  sSIG[16][V];
  __shared__ __align__(16) __bf16 sFV[3][16][V + PADB];
  __shared__ float  sFVD[16][V];
  __shared__ __align__(16) __bf16 sS2[16][STP + PADB];
  __shared__ __align__(16) __bf16 sV2[3][16][VTP + PADB];
  __shared__ float  sALPHA[16][HEADS];

  const int tid = threadIdx.x;
  const int e0 = blockIdx.x * 16;
  if (tid < 16) {
    const int e = e0 + tid;
    sSrc[tid] = edge_src[e];
    sDst[tid] = edge_dst[e];
    sY0[tid] = edge_attr[e * 4 + 0];
    sY1[tid][0] = edge_attr[e * 4 + 1];
    sY1[tid][1] = edge_attr[e * 4 + 2];
    sY1[tid][2] = edge_attr[e * 4 + 3];
    // prefetch next tile's per-edge rows (global_prefetch_b8)
    if (e0 + 16 < nEdges) {
      __builtin_prefetch(&edge_emb[(e0 + 16 + tid) * NB], 0, 1);
      __builtin_prefetch(&edge_attr[(e0 + 16 + tid) * 4], 0, 1);
    }
  }
  __syncthreads();
  // gather edge scalar/vector features and radial MLP input
  for (int idx = tid; idx < 16 * S; idx += 256) {
    const int i = idx >> 7, c = idx & (S - 1);
    sES[i][c] = ms[sSrc[i] * S + c] + dsb[sDst[i] * S + c];
  }
  for (int idx = tid; idx < 16 * V * 3; idx += 256) {
    const int i = idx / (V * 3), r = idx % (V * 3);
    const int comp = r / V, c = r % V;
    sEV[i][c][comp] = mv[sSrc[i] * (3 * V) + comp * V + c] +
                      dvv[sDst[i] * (3 * V) + comp * V + c];
  }
  for (int idx = tid; idx < 16 * FIN; idx += 256) {
    const int i = idx / FIN, c = idx % FIN;
    float val;
    if (c < NB)             val = edge_emb[(e0 + i) * NB + c];
    else if (c < NB + ATTR) val = node_attr[sSrc[i] * ATTR + (c - NB)];
    else                    val = node_attr[sDst[i] * ATTR + (c - NB - ATTR)];
    sWF[i][c] = (__bf16)val;
  }
  __syncthreads();
  for (int idx = tid; idx < 16 * V; idx += 256) {
    const int i = idx >> 6, c = idx & (V - 1);
    sEVD[i][c] = sEV[i][c][0] * sY1[i][0] + sEV[i][c][1] * sY1[i][1] +
                 sEV[i][c][2] * sY1[i][2];
  }
  const int lane = tid & 31, l15 = lane & 15, hf = lane >> 4;
  const int wv = tid >> 5;
  // radial MLP layer 0 (160->64) with SiLU
  for (int t = wv; t < 4; t += 8) {
    v8f acc = {};
    acc = wmma_gemm16(&sWF[0][0], FIN + PADB, tradW0, FIN, t * 16, FIN, acc);
    const float b = radb0[t * 16 + l15];
#pragma unroll
    for (int r = 0; r < 8; ++r) {
      const float x = acc[r] + b;
      sH1[r + hf * 8][t * 16 + l15] = (__bf16)(x * sigf(x));
    }
  }
  __syncthreads();
  // layer 1 (64->64) with SiLU
  for (int t = wv; t < 4; t += 8) {
    v8f acc = {};
    acc = wmma_gemm16(&sH1[0][0], FC + PADB, tradW1, FC, t * 16, FC, acc);
    const float b = radb1[t * 16 + l15];
#pragma unroll
    for (int r = 0; r < 8; ++r) {
      const float x = acc[r] + b;
      sH2[r + hf * 8][t * 16 + l15] = (__bf16)(x * sigf(x));
    }
  }
  __syncthreads();
  // layer 2 (64->448); pass1 only needs w_ss (cols 0..127) and w_dt (320..383)
  for (int t = wv; t < 28; t += 8) {
    if (!FULL && !(t < 8 || (t >= 20 && t < 24))) continue;
    v8f acc = {};
    acc = wmma_gemm16(&sH2[0][0], FC + PADB, tradW2, FC, t * 16, FC, acc);
    const float b = radb2[t * 16 + l15];
#pragma unroll
    for (int r = 0; r < 8; ++r)
      sW[r + hf * 8][t * 16 + l15] = (__bf16)(acc[r] + b);
  }
  __syncthreads();
  // depthwise tensor product 1 -> s_tp (and v_tp for FULL)
  for (int idx = tid; idx < 16 * STP; idx += 256) {
    const int i = idx / STP, j = idx % STP;
    float val;
    if (j < S) val = (float)sW[i][j] * sES[i][j] * sY0[i];
    else { const int c = j - S; val = (float)sW[i][320 + c] * sEVD[i][c]; }
    sSTP[i][j] = (__bf16)val;
  }
  if (FULL) {
    for (int idx = tid; idx < 3 * 16 * VTP; idx += 256) {
      const int comp = idx / (16 * VTP);
      const int rem = idx % (16 * VTP);
      const int i = rem / VTP, k = rem % VTP;
      float val;
      if (k < S) {
        val = (float)sW[i][S + k] * sES[i][k] * sY1[i][comp];
      } else if (k < S + V) {
        const int c = k - S;
        val = (float)sW[i][256 + c] * sEV[i][c][comp] * sY0[i];
      } else {
        const int c = k - S - V;
        const int c1 = (comp + 1) % 3, c2 = (comp + 2) % 3;
        const float cr = sEV[i][c][c1] * sY1[i][c2] - sEV[i][c][c2] * sY1[i][c1];
        val = (float)sW[i][384 + c] * cr;
      }
      sVTP[comp][i][k] = (__bf16)val;
    }
  }
  __syncthreads();

  if (!FULL) {
    // attention scores: sc = s_tp @ att_Ws + att_bs, smooth-leaky, head dots
    for (int t = wv; t < 8; t += 8) {
      v8f acc = {};
      acc = wmma_gemm16(&sSTP[0][0], STP + PADB, tattWs, STP, t * 16, STP, acc);
      const float b = attbs[t * 16 + l15];
#pragma unroll
      for (int r = 0; r < 8; ++r) {
        const float x = acc[r] + b;
        const float sl = 0.6f * x + 0.4f * x * (2.0f * sigf(x) - 1.0f);
        sSC[r + hf * 8][t * 16 + l15] = sl;
      }
    }
    __syncthreads();
    if (tid < 16 * HEADS) {
      const int i = tid >> 2, h = tid & 3;
      float acc = 0.f;
#pragma unroll
      for (int k = 0; k < SHD; ++k) acc += sSC[i][h * SHD + k] * attdot[h * SHD + k];
      logits[(e0 + i) * HEADS + h] = acc;
    }
    return;
  }

  // gate linear: gs = s_tp @ lin1_Ws + bs ; fs = silu(gs[:128]), sig = sigmoid(gs[128:])
  for (int t = wv; t < 12; t += 8) {
    v8f acc = {};
    acc = wmma_gemm16(&sSTP[0][0], STP + PADB, tlin1Ws, STP, t * 16, STP, acc);
    const int col = t * 16 + l15;
    const float b = lin1bs[col];
#pragma unroll
    for (int r = 0; r < 8; ++r) {
      const float x = acc[r] + b;
      const int row = r + hf * 8;
      if (col < S) sFS[row][col] = (__bf16)(x * sigf(x));
      else         sSIG[row][col - S] = sigf(x);
    }
  }
  __syncthreads();
  // gv = v_tp @ lin1_Wv, gated by sig -> fv
  for (int j = wv; j < 12; j += 8) {
    const int comp = j >> 2, t = j & 3;
    v8f acc = {};
    acc = wmma_gemm16(&sVTP[comp][0][0], VTP + PADB, tlin1Wv, VTP, t * 16, VTP, acc);
    const int col = t * 16 + l15;
#pragma unroll
    for (int r = 0; r < 8; ++r) {
      const int row = r + hf * 8;
      sFV[comp][row][col] = (__bf16)(acc[r] * sSIG[row][col]);
    }
  }
  __syncthreads();
  for (int idx = tid; idx < 16 * V; idx += 256) {
    const int i = idx >> 6, c = idx & (V - 1);
    sFVD[i][c] = (float)sFV[0][i][c] * sY1[i][0] + (float)sFV[1][i][c] * sY1[i][1] +
                 (float)sFV[2][i][c] * sY1[i][2];
  }
  __syncthreads();
  // depthwise tensor product 2 with shared weights tp2_w
  for (int idx = tid; idx < 16 * STP; idx += 256) {
    const int i = idx / STP, j = idx % STP;
    float val;
    if (j < S) val = tp2w[j] * (float)sFS[i][j] * sY0[i];
    else { const int c = j - S; val = tp2w[320 + c] * sFVD[i][c]; }
    sS2[i][j] = (__bf16)val;
  }
  for (int idx = tid; idx < 3 * 16 * VTP; idx += 256) {
    const int comp = idx / (16 * VTP);
    const int rem = idx % (16 * VTP);
    const int i = rem / VTP, k = rem % VTP;
    float val;
    if (k < S) {
      val = tp2w[S + k] * (float)sFS[i][k] * sY1[i][comp];
    } else if (k < S + V) {
      const int c = k - S;
      val = tp2w[256 + c] * (float)sFV[comp][i][c] * sY0[i];
    } else {
      const int c = k - S - V;
      const int c1 = (comp + 1) % 3, c2 = (comp + 2) % 3;
      const float cr = (float)sFV[c1][i][c] * sY1[i][c2] -
                       (float)sFV[c2][i][c] * sY1[i][c1];
      val = tp2w[384 + c] * cr;
    }
    sV2[comp][i][k] = (__bf16)val;
  }
  if (tid < 16 * HEADS) {
    const int i = tid >> 2, h = tid & 3;
    const int d = sDst[i];
    const float l = logits[(e0 + i) * HEADS + h];
    sALPHA[i][h] = expf(l - nmax[d * HEADS + h]) / (nden[d * HEADS + h] + 1e-12f);
  }
  __syncthreads();
  // value projections + alpha-weighted scatter into node accumulators
  for (int j = wv; j < 20; j += 8) {
    if (j < 8) {
      v8f acc = {};
      acc = wmma_gemm16(&sS2[0][0], STP + PADB, tlin2Ws, STP, j * 16, STP, acc);
      const int col = j * 16 + l15;
      const float b = lin2bs[col];
      const int h = col >> 5;
#pragma unroll
      for (int r = 0; r < 8; ++r) {
        const int row = r + hf * 8;
        atomicAdd(&acc_s[sDst[row] * S + col], (acc[r] + b) * sALPHA[row][h]);
      }
    } else {
      const int jj = j - 8, comp = jj >> 2, t = jj & 3;
      v8f acc = {};
      acc = wmma_gemm16(&sV2[comp][0][0], VTP + PADB, tlin2Wv, VTP, t * 16, VTP, acc);
      const int col = t * 16 + l15;
      const int h = col >> 4;
#pragma unroll
      for (int r = 0; r < 8; ++r) {
        const int row = r + hf * 8;
        atomicAdd(&acc_v[sDst[row] * (3 * V) + col * 3 + comp],
                  acc[r] * sALPHA[row][h]);
      }
    }
  }
}

// ---------------- softmax segment reductions ----------------
__global__ void init_kernel(float* nmax, float* nden, float* acc_s, float* acc_v) {
  const int nNH = nNodes * HEADS;
  const int total = 2 * nNH + nNodes * S + nNodes * 3 * V;
  for (int i = blockIdx.x * blockDim.x + threadIdx.x; i < total;
       i += gridDim.x * blockDim.x) {
    if (i < nNH)               nmax[i] = -3.0e38f;
    else if (i < 2 * nNH)      nden[i - nNH] = 0.f;
    else if (i < 2 * nNH + nNodes * S) acc_s[i - 2 * nNH] = 0.f;
    else                       acc_v[i - 2 * nNH - nNodes * S] = 0.f;
  }
}

__global__ void segmax_kernel(const float* __restrict__ logits,
                              const int* __restrict__ edge_dst, float* nmax) {
  const int idx = blockIdx.x * blockDim.x + threadIdx.x;
  if (idx >= nEdges * HEADS) return;
  const int e = idx >> 2, h = idx & 3;
  atomicMaxFloat(&nmax[edge_dst[e] * HEADS + h], logits[idx]);
}

__global__ void segden_kernel(const float* __restrict__ logits,
                              const int* __restrict__ edge_dst,
                              const float* __restrict__ nmax, float* nden) {
  const int idx = blockIdx.x * blockDim.x + threadIdx.x;
  if (idx >= nEdges * HEADS) return;
  const int e = idx >> 2, h = idx & 3;
  const int d = edge_dst[e];
  atomicAdd(&nden[d * HEADS + h], expf(logits[idx] - nmax[d * HEADS + h]));
}

// ---------------- final node output linears ----------------
__global__ __launch_bounds__(256) void node_out_kernel(
    const float* __restrict__ acc_s, const float* __restrict__ acc_v,
    const __bf16* __restrict__ toutWs, const float* __restrict__ outbs,
    const __bf16* __restrict__ toutWv, float* __restrict__ out) {
  __shared__ __align__(16) __bf16 sS_[16][S + PADB];
  __shared__ __align__(16) __bf16 sV_[3][16][V + PADB];
  const int n0 = blockIdx.x * 16;
  for (int idx = threadIdx.x; idx < 16 * S; idx += 256) {
    const int i = idx >> 7, c = idx & (S - 1);
    sS_[i][c] = (__bf16)acc_s[(n0 + i) * S + c];
  }
  for (int idx = threadIdx.x; idx < 16 * 3 * V; idx += 256) {
    const int i = idx / (3 * V), r = idx % (3 * V);
    const int c = r / 3, comp = r % 3;
    sV_[comp][i][c] = (__bf16)acc_v[(n0 + i) * (3 * V) + c * 3 + comp];
  }
  __syncthreads();
  const int lane = threadIdx.x & 31, l15 = lane & 15, hf = lane >> 4;
  for (int job = threadIdx.x >> 5; job < 20; job += 8) {
    v8f acc = {};
    if (job < 8) {
      const int c0 = job * 16;
      acc = wmma_gemm16(&sS_[0][0], S + PADB, toutWs, S, c0, S, acc);
      const float b = outbs[c0 + l15];
#pragma unroll
      for (int r = 0; r < 8; ++r)
        out[(n0 + r + hf * 8) * 320 + c0 + l15] = acc[r] + b;
    } else {
      const int j = job - 8, comp = j >> 2, c0 = (j & 3) * 16;
      acc = wmma_gemm16(&sV_[comp][0][0], V + PADB, toutWv, V, c0, V, acc);
#pragma unroll
      for (int r = 0; r < 8; ++r)
        out[(n0 + r + hf * 8) * 320 + S + (c0 + l15) * 3 + comp] = acc[r];
    }
  }
}

// ---------------- host side ----------------
extern "C" void kernel_launch(void* const* d_in, const int* in_sizes, int n_in,
                              void* d_out, int out_size, void* d_ws, size_t ws_size,
                              hipStream_t stream) {
  (void)in_sizes; (void)n_in; (void)out_size; (void)ws_size;
  const float* node_input = (const float*)d_in[0];
  const float* node_attr  = (const float*)d_in[1];
  const float* edge_attr  = (const float*)d_in[2];
  const float* edge_emb   = (const float*)d_in[3];
  const int*   edge_src   = (const int*)d_in[4];
  const int*   edge_dst   = (const int*)d_in[5];
  const float* Wss   = (const float*)d_in[6];
  const float* bss   = (const float*)d_in[7];
  const float* Wsv   = (const float*)d_in[8];
  const float* Wds   = (const float*)d_in[9];
  const float* Wdv   = (const float*)d_in[10];
  const float* radW0 = (const float*)d_in[11];
  const float* radb0 = (const float*)d_in[12];
  const float* radW1 = (const float*)d_in[13];
  const float* radb1 = (const float*)d_in[14];
  const float* radW2 = (const float*)d_in[15];
  const float* radb2 = (const float*)d_in[16];
  const float* lin1Ws = (const float*)d_in[17];
  const float* lin1bs = (const float*)d_in[18];
  const float* lin1Wv = (const float*)d_in[19];
  const float* attWs  = (const float*)d_in[20];
  const float* attbs  = (const float*)d_in[21];
  const float* tp2w   = (const float*)d_in[22];
  const float* lin2Ws = (const float*)d_in[23];
  const float* lin2bs = (const float*)d_in[24];
  const float* lin2Wv = (const float*)d_in[25];
  const float* attdot = (const float*)d_in[26];
  const float* outWs  = (const float*)d_in[27];
  const float* outbs  = (const float*)d_in[28];
  const float* outWv  = (const float*)d_in[29];
  float* out = (float*)d_out;

  float* ws = (float*)d_ws;
  float* ms     = ws; ws += nNodes * S;
  float* dsb    = ws; ws += nNodes * S;
  float* mv     = ws; ws += nNodes * 3 * V;
  float* dvv    = ws; ws += nNodes * 3 * V;
  float* logits = ws; ws += nEdges * HEADS;
  float* nmax   = ws; ws += nNodes * HEADS;
  float* nden   = ws; ws += nNodes * HEADS;
  float* acc_s  = ws; ws += nNodes * S;
  float* acc_v  = ws; ws += nNodes * 3 * V;

  // bf16 transposed-weight area (starts 64B-aligned: 43,840,000 bytes in)
  __bf16* bws = (__bf16*)ws;
  __bf16* tWss    = bws; bws += S * S;
  __bf16* tWds    = bws; bws += S * S;
  __bf16* tWsv    = bws; bws += V * V;
  __bf16* tWdv    = bws; bws += V * V;
  __bf16* tradW0  = bws; bws += FIN * FC;
  __bf16* tradW1  = bws; bws += FC * FC;
  __bf16* tradW2  = bws; bws += FC * NW;
  __bf16* tattWs  = bws; bws += STP * S;
  __bf16* tlin1Ws = bws; bws += STP * STP;
  __bf16* tlin1Wv = bws; bws += VTP * V;
  __bf16* tlin2Ws = bws; bws += STP * S;
  __bf16* tlin2Wv = bws; bws += VTP * V;
  __bf16* toutWs  = bws; bws += S * S;
  __bf16* toutWv  = bws; bws += V * V;

  auto wc = [&](const float* src, __bf16* dst, int K, int N) {
    wconv_kernel<<<(K * N + 255) / 256, 256, 0, stream>>>(src, dst, K, N);
  };
  wc(Wss, tWss, S, S);        wc(Wds, tWds, S, S);
  wc(Wsv, tWsv, V, V);        wc(Wdv, tWdv, V, V);
  wc(radW0, tradW0, FIN, FC); wc(radW1, tradW1, FC, FC);
  wc(radW2, tradW2, FC, NW);  wc(attWs, tattWs, STP, S);
  wc(lin1Ws, tlin1Ws, STP, STP); wc(lin1Wv, tlin1Wv, VTP, V);
  wc(lin2Ws, tlin2Ws, STP, S);   wc(lin2Wv, tlin2Wv, VTP, V);
  wc(outWs, toutWs, S, S);       wc(outWv, toutWv, V, V);

  node_proj_kernel<<<nNodes / 16, 256, 0, stream>>>(
      node_input, tWss, bss, tWsv, tWds, tWdv, ms, dsb, mv, dvv);
  init_kernel<<<4096, 256, 0, stream>>>(nmax, nden, acc_s, acc_v);
  edge_kernel<false><<<nEdges / 16, 256, 0, stream>>>(
      ms, dsb, mv, dvv, edge_src, edge_dst, edge_attr, edge_emb, node_attr,
      tradW0, radb0, tradW1, radb1, tradW2, radb2, tattWs, attbs, attdot,
      tlin1Ws, lin1bs, tlin1Wv, tp2w, tlin2Ws, lin2bs, tlin2Wv,
      logits, nmax, nden, acc_s, acc_v);
  segmax_kernel<<<(nEdges * HEADS + 255) / 256, 256, 0, stream>>>(logits, edge_dst, nmax);
  segden_kernel<<<(nEdges * HEADS + 255) / 256, 256, 0, stream>>>(logits, edge_dst, nmax, nden);
  edge_kernel<true><<<nEdges / 16, 256, 0, stream>>>(
      ms, dsb, mv, dvv, edge_src, edge_dst, edge_attr, edge_emb, node_attr,
      tradW0, radb0, tradW1, radb1, tradW2, radb2, tattWs, attbs, attdot,
      tlin1Ws, lin1bs, tlin1Wv, tp2w, tlin2Ws, lin2bs, tlin2Wv,
      logits, nmax, nden, acc_s, acc_v);
  node_out_kernel<<<nNodes / 16, 256, 0, stream>>>(
      acc_s, acc_v, toutWs, outbs, toutWv, out);
}
